// GeneratorSelfAttention_76201309765722
// MI455X (gfx1250) — compile-verified, matrix operand order checked by reading, not run on hardware
//
#include <hip/hip_runtime.h>
#include <hip/hip_bf16.h>

// ---------------------------------------------------------------------------
// GATr equivariant self-attention, fused for MI455X (gfx1250, wave32, WMMA).
// All heavy math runs through v_wmma_f32_16x16x32_bf16 (fp32 accumulate).
// ---------------------------------------------------------------------------

typedef __bf16 v16bf __attribute__((ext_vector_type(16)));
typedef float  v8f   __attribute__((ext_vector_type(8)));
typedef unsigned int u32x4 __attribute__((ext_vector_type(4)));

union Frag { u32x4 q[2]; v16bf v; };

__device__ __forceinline__ unsigned short f2bf(float x) {
  union { float f; unsigned u; } c; c.f = x;
  unsigned r = c.u + 0x7FFFu + ((c.u >> 16) & 1u);   // round-to-nearest-even
  return (unsigned short)(r >> 16);
}
__device__ __forceinline__ float bf2f(unsigned short s) {
  union { float f; unsigned u; } c; c.u = ((unsigned)s) << 16; return c.f;
}

// Basis structure tables (from make_basis): grade of each component, e0 map
// source component, e0 basis index, and INV-feature index.
__constant__ int d_grade[16] = {0,1,1,1,1,2,2,2,2,2,2,3,3,3,3,4};
__constant__ int d_src[16]   = {0,0,2,3,4,2,3,4,8,9,10,8,9,10,14,14};
__constant__ int d_aidx[16]  = {0,5,0,0,0,6,6,6,0,0,0,7,7,7,0,8};
__constant__ int d_isdst[16] = {0,1,0,0,0,1,1,1,0,0,0,1,1,1,0,1};
__constant__ int d_finv[16]  = {0,-1,1,2,3,-1,-1,-1,4,5,6,-1,-1,-1,7,-1};

#define BB 4
#define TT 1024
#define CC 48
#define HH 8
#define BT 4096
#define BH 32
#define DQK 544      // 528 used (384 inv + 144 plane) + 16 pad
#define DV  768      // C*16 v columns per head
#define INVS 5.1031036e-2f   // 1/sqrt(8*C) = 1/sqrt(384)

// ---------------------------------------------------------------------------
// Weight prep: effective bf16 weights per output component j.
// W1eff[j][o][k], k in [0,128): [0,48)=grade part, [64,112)=e0 part, rest 0.
// ---------------------------------------------------------------------------
__global__ void prep_w1(const float* __restrict__ wa, unsigned short* __restrict__ w1) {
  long idx = (long)blockIdx.x * 256 + threadIdx.x;
  if (idx >= 16L * 1152 * 128) return;
  int j = (int)(idx / (1152 * 128));
  int r = (int)(idx % (1152 * 128));
  int o = r / 128, k = r % 128;
  float v = 0.0f;
  if (k < 48)                              v = wa[(o * 48 + k) * 9 + d_grade[j]];
  else if (k >= 64 && k < 112 && d_isdst[j]) v = wa[(o * 48 + (k - 64)) * 9 + d_aidx[j]];
  w1[idx] = f2bf(v);
}

// W2eff[j][o][k], k in [0,768): [0,384)=grade part, [384,768)=e0 part.
__global__ void prep_w2(const float* __restrict__ wn, unsigned short* __restrict__ w2) {
  long idx = (long)blockIdx.x * 256 + threadIdx.x;
  if (idx >= 16L * 48 * 768) return;
  int j = (int)(idx / (48 * 768));
  int r = (int)(idx % (48 * 768));
  int o = r / 768, k = r % 768;
  float v;
  if (k < 384) v = wn[(o * 384 + k) * 9 + d_grade[j]];
  else         v = d_isdst[j] ? wn[(o * 384 + (k - 384)) * 9 + d_aidx[j]] : 0.0f;
  w2[idx] = f2bf(v);
}

// ---------------------------------------------------------------------------
// Fused equi-RMS-norm -> bf16, component-major: hbf[bt][m][64] (cols 48..63=0)
// ---------------------------------------------------------------------------
__global__ void norm_kernel(const float* __restrict__ hid, const float* __restrict__ lnw,
                            unsigned short* __restrict__ hbf) {
  int bt = blockIdx.x, tid = threadIdx.x;
  __shared__ float red[128];
  const float* x = hid + (size_t)bt * 768;
  float s = 0.0f;
  for (int i = tid; i < 768; i += 128) {
    int m = i & 15;
    if ((0x471D >> m) & 1) { float v = x[i]; s += v * v; }   // INV mask
  }
  red[tid] = s; __syncthreads();
  for (int off = 64; off > 0; off >>= 1) {
    if (tid < off) red[tid] += red[tid + off];
    __syncthreads();
  }
  float rs = rsqrtf(red[0] * (1.0f / 48.0f) + 1e-6f);
  for (int i = tid; i < 16 * 64; i += 128) {
    int m = i >> 6, c = i & 63;
    float v = (c < 48) ? x[c * 16 + m] * rs * lnw[c] : 0.0f;
    hbf[(size_t)bt * 1024 + i] = f2bf(v);
  }
}

// ---------------------------------------------------------------------------
// QKV projection GEMM (per component j): D = H(16x128) x W1^T, WMMA bf16.
// Scatters results straight into attention-ready buffers:
//   q_feat/k_feat (B*H, T, 544) bf16, pre-scaled; vT (B*H, 768, T) bf16.
// ---------------------------------------------------------------------------
__global__ void qkv_gemm(const unsigned short* __restrict__ hbf,
                         const unsigned short* __restrict__ w1,
                         const float* __restrict__ b_attn,
                         unsigned short* __restrict__ qf,
                         unsigned short* __restrict__ kf,
                         unsigned short* __restrict__ vt) {
  int j = blockIdx.z, tbase = blockIdx.y * 16, obase = blockIdx.x * 16;
  int lane = threadIdx.x & 31, n = lane & 15, hi = (lane >> 4) & 1;
  int srcj = d_src[j];
  v8f acc = {};
  for (int ch = 0; ch < 4; ++ch) {
    int slice = (ch < 2) ? j : srcj;
    int koff = (ch & 1) * 32;
    Frag a, b;
    const unsigned short* pa = hbf + ((size_t)(tbase + n) * 16 + slice) * 64 + koff + hi * 8;
    a.q[0] = *(const u32x4*)pa; a.q[1] = *(const u32x4*)(pa + 16);
    const unsigned short* pb = w1 + ((size_t)j * 1152 + obase + n) * 128 + ch * 32 + hi * 8;
    b.q[0] = *(const u32x4*)pb; b.q[1] = *(const u32x4*)(pb + 16);
    acc = __builtin_amdgcn_wmma_f32_16x16x32_bf16(false, a.v, false, b.v, (short)0, acc, false, false);
  }
  int o = obase + n;
  int which = o / 384, hc = o % 384, h = hc / 48, c = hc % 48;
  float bias = (j == 0) ? b_attn[o] : 0.0f;
  for (int g = 0; g < 8; ++g) {
    int t = tbase + g + hi * 8;
    int b0 = t >> 10, t0 = t & 1023;
    float v = acc[g] + bias;
    size_t row = (size_t)(b0 * 8 + h) * 1024 + t0;
    if (which == 2) {
      vt[((size_t)(b0 * 8 + h) * 768 + c * 16 + j) * 1024 + t0] = f2bf(v);
    } else {
      unsigned short* dst = (which == 0) ? qf : kf;
      int fi = d_finv[j];
      if (fi >= 0) {                         // INV feature
        dst[row * DQK + c * 8 + fi] = f2bf(which == 0 ? v * INVS : v);
      } else if (j >= 11 && j <= 13) {       // plane feature (2x on q side)
        dst[row * DQK + 384 + c * 3 + (j - 11)] = f2bf(which == 0 ? v * 2.0f * INVS : v);
      }                                      // components 1,5,6,7,15 unused by q/k
    }
  }
}

// kn[s] = |pk|^2 per key (column bias for logits; qn cancels in softmax)
__global__ void kn_kernel(const unsigned short* __restrict__ kf, float* __restrict__ kn) {
  int idx = blockIdx.x * 256 + threadIdx.x;
  if (idx >= BH * TT) return;
  const unsigned short* p = kf + (size_t)idx * DQK + 384;
  float s = 0.0f;
  for (int i = 0; i < 144; ++i) { float v = bf2f(p[i]); s += v * v; }
  kn[idx] = s;
}

// ---------------------------------------------------------------------------
// Flash attention: one WG = 16 queries of one (b,h). 4 waves; key blocks of
// 256 with online softmax. Each wave owns 192 V-columns (12 accum tiles).
// ---------------------------------------------------------------------------
__global__ __launch_bounds__(128) void attn_kernel(const unsigned short* __restrict__ qf,
                                                   const unsigned short* __restrict__ kf,
                                                   const unsigned short* __restrict__ vt,
                                                   const float* __restrict__ kn,
                                                   unsigned short* __restrict__ a2) {
  int qt = blockIdx.x, bh = blockIdx.y;
  int tid = threadIdx.x, w = tid >> 5, lane = tid & 31;
  int n = lane & 15, hi = (lane >> 4) & 1;

  __shared__ alignas(16) unsigned int qtile[16 * 272];     // 16 x 544 bf16
  __shared__ alignas(16) float Ls[16 * 256];
  __shared__ alignas(16) unsigned short Pb[16 * 256];
  __shared__ float s_m[16], s_l[16], s_f[16];
  __shared__ float red[16][8];

  const unsigned int* qsrc = (const unsigned int*)(qf + ((size_t)bh * TT + qt * 16) * DQK);
  for (int i = tid; i < 16 * 272; i += 128) qtile[i] = qsrc[i];
  if (tid < 16) { s_m[tid] = -1e30f; s_l[tid] = 0.0f; }
  __syncthreads();

  v8f acc[12] = {};

  for (int kb = 0; kb < 4; ++kb) {
    // ---- logits: Q(16x544) x K^T block(544x256), split over 4 waves ----
    for (int sti = 0; sti < 4; ++sti) {
      int st = w * 4 + sti;
      int s0 = kb * 256 + st * 16;
      v8f lg = {};
      const unsigned short* qrow = (const unsigned short*)qtile + n * DQK;
      const unsigned short* krow = kf + ((size_t)bh * TT + s0 + n) * DQK;
      for (int ch = 0; ch < 17; ++ch) {
        Frag a, b;
        const unsigned short* pa = qrow + ch * 32 + hi * 8;
        a.q[0] = *(const u32x4*)pa; a.q[1] = *(const u32x4*)(pa + 16);
        const unsigned short* pb = krow + ch * 32 + hi * 8;
        b.q[0] = *(const u32x4*)pb; b.q[1] = *(const u32x4*)(pb + 16);
        lg = __builtin_amdgcn_wmma_f32_16x16x32_bf16(false, a.v, false, b.v, (short)0, lg, false, false);
      }
      float bias = -kn[(size_t)bh * TT + s0 + n] * INVS;
      for (int g = 0; g < 8; ++g)
        Ls[(g + hi * 8) * 256 + st * 16 + n] = lg[g] + bias;
    }
    __syncthreads();

    // ---- online softmax over the 256-key block ----
    int r = tid >> 3, sub = tid & 7;
    float mx = -1e30f;
    for (int ci = 0; ci < 32; ++ci) mx = fmaxf(mx, Ls[r * 256 + sub * 32 + ci]);
    red[r][sub] = mx; __syncthreads();
    if (sub == 0) {
      float bm = red[r][0];
      for (int i = 1; i < 8; ++i) bm = fmaxf(bm, red[r][i]);
      float mn = fmaxf(s_m[r], bm);
      s_f[r] = __expf(s_m[r] - mn);
      s_m[r] = mn;
    }
    __syncthreads();
    float mrow = s_m[r], sm = 0.0f;
    for (int ci = 0; ci < 32; ++ci) {
      float p = __expf(Ls[r * 256 + sub * 32 + ci] - mrow);
      Pb[r * 256 + sub * 32 + ci] = f2bf(p);
      sm += p;
    }
    red[r][sub] = sm; __syncthreads();
    if (sub == 0) {
      float tot = 0.0f;
      for (int i = 0; i < 8; ++i) tot += red[r][i];
      s_l[r] = s_l[r] * s_f[r] + tot;
    }
    __syncthreads();

    // ---- PV: rescale accumulators, then P(16x256) x V(256x192) per wave ----
    for (int ct = 0; ct < 12; ++ct)
      for (int g = 0; g < 8; ++g)
        acc[ct][g] *= s_f[g + hi * 8];
    int colbase = w * 192;
    for (int sc = 0; sc < 8; ++sc) {
      Frag a;
      const unsigned short* pa = Pb + n * 256 + sc * 32 + hi * 8;
      a.q[0] = *(const u32x4*)pa; a.q[1] = *(const u32x4*)(pa + 16);
      for (int ct = 0; ct < 12; ++ct) {
        Frag b;
        const unsigned short* pb = vt + ((size_t)bh * DV + colbase + ct * 16 + n) * TT
                                      + kb * 256 + sc * 32 + hi * 8;
        b.q[0] = *(const u32x4*)pb; b.q[1] = *(const u32x4*)(pb + 16);
        acc[ct] = __builtin_amdgcn_wmma_f32_16x16x32_bf16(false, a.v, false, b.v, (short)0, acc[ct], false, false);
      }
    }
    __syncthreads();
  }

  // ---- epilogue: normalize, write component-major bf16 (BT, 16, 384) ----
  int b0 = bh >> 3, h0 = bh & 7;
  for (int ct = 0; ct < 12; ++ct) {
    int vcol = w * 192 + ct * 16 + n;
    int j = vcol & 15, c = vcol >> 4;
    int ii = h0 * 48 + c;
    for (int g = 0; g < 8; ++g) {
      int row = g + hi * 8;
      float v = acc[ct][g] / s_l[row];
      int t = qt * 16 + row;
      a2[((size_t)(b0 * 1024 + t) * 16 + j) * 384 + ii] = f2bf(v);
    }
  }
}

// ---------------------------------------------------------------------------
// Output projection (16 GEMMs, K=768) + bias + residual -> d_out f32
// ---------------------------------------------------------------------------
__global__ void proj2_kernel(const unsigned short* __restrict__ a2,
                             const unsigned short* __restrict__ w2,
                             const float* __restrict__ b_next,
                             const float* __restrict__ hid,
                             float* __restrict__ out) {
  int j = blockIdx.z, tbase = blockIdx.y * 16, obase = blockIdx.x * 16;
  int lane = threadIdx.x & 31, n = lane & 15, hi = (lane >> 4) & 1;
  int srcj = d_src[j];
  v8f acc = {};
  for (int ch = 0; ch < 24; ++ch) {
    int slice = (ch < 12) ? j : srcj;
    int koff = (ch % 12) * 32;
    Frag a, b;
    const unsigned short* pa = a2 + ((size_t)(tbase + n) * 16 + slice) * 384 + koff + hi * 8;
    a.q[0] = *(const u32x4*)pa; a.q[1] = *(const u32x4*)(pa + 16);
    const unsigned short* pb = w2 + ((size_t)j * 48 + obase + n) * 768 + ch * 32 + hi * 8;
    b.q[0] = *(const u32x4*)pb; b.q[1] = *(const u32x4*)(pb + 16);
    acc = __builtin_amdgcn_wmma_f32_16x16x32_bf16(false, a.v, false, b.v, (short)0, acc, false, false);
  }
  int o = obase + n;
  float bias = (j == 0) ? b_next[o] : 0.0f;
  for (int g = 0; g < 8; ++g) {
    int t = tbase + g + hi * 8;
    size_t oi = (size_t)t * 768 + o * 16 + j;
    out[oi] = acc[g] + bias + hid[oi];
  }
}

// ---------------------------------------------------------------------------
// Workspace layout (bytes, 256-aligned). Total ~186.4 MB.
// ---------------------------------------------------------------------------
static const size_t OFF_HBF = 0;                       // 4096*16*64*2   = 8,388,608
static const size_t OFF_W1  = 8388608;                 // 16*1152*128*2  = 4,718,592
static const size_t OFF_W2  = 13107200;                // 16*48*768*2    = 1,179,648
static const size_t OFF_QF  = 14286848;                // 32*1024*544*2  = 35,651,584
static const size_t OFF_KF  = 49938432;                // 35,651,584
static const size_t OFF_VT  = 85590016;                // 32*768*1024*2  = 50,331,648
static const size_t OFF_KN  = 135921664;               // 32768*4        = 131,072
static const size_t OFF_A2  = 136052736;               // 4096*16*384*2  = 50,331,648

extern "C" void kernel_launch(void* const* d_in, const int* in_sizes, int n_in,
                              void* d_out, int out_size, void* d_ws, size_t ws_size,
                              hipStream_t stream) {
  (void)in_sizes; (void)n_in; (void)out_size; (void)ws_size;
  const float* hidden = (const float*)d_in[0];
  const float* w_attn = (const float*)d_in[1];
  const float* b_attn = (const float*)d_in[2];
  const float* w_next = (const float*)d_in[3];
  const float* b_next = (const float*)d_in[4];
  const float* ln_w   = (const float*)d_in[5];
  // d_in[6] (basis) is a fixed structure; encoded in constant tables above.

  char* ws = (char*)d_ws;
  unsigned short* hbf = (unsigned short*)(ws + OFF_HBF);
  unsigned short* w1  = (unsigned short*)(ws + OFF_W1);
  unsigned short* w2  = (unsigned short*)(ws + OFF_W2);
  unsigned short* qfb = (unsigned short*)(ws + OFF_QF);
  unsigned short* kfb = (unsigned short*)(ws + OFF_KF);
  unsigned short* vtb = (unsigned short*)(ws + OFF_VT);
  float*          knb = (float*)(ws + OFF_KN);
  unsigned short* a2b = (unsigned short*)(ws + OFF_A2);

  // zero feature buffers (pad columns must stay zero for the K=544 GEMM)
  hipMemsetAsync(qfb, 0, (size_t)BH * TT * DQK * 2, stream);
  hipMemsetAsync(kfb, 0, (size_t)BH * TT * DQK * 2, stream);

  prep_w1<<<9216, 256, 0, stream>>>(w_attn, w1);
  prep_w2<<<2304, 256, 0, stream>>>(w_next, w2);
  norm_kernel<<<BT, 128, 0, stream>>>(hidden, ln_w, hbf);
  qkv_gemm<<<dim3(72, 256, 16), 32, 0, stream>>>(hbf, w1, b_attn, qfb, kfb, vtb);
  kn_kernel<<<128, 256, 0, stream>>>(kfb, knb);
  attn_kernel<<<dim3(64, 32), 128, 0, stream>>>(qfb, kfb, vtb, knb, a2b);
  proj2_kernel<<<dim3(3, 256, 16), 32, 0, stream>>>(a2b, w2, b_next, hidden, (float*)d_out);
}
